// GATNet_4312147165188
// MI455X (gfx1250) — matrix operand order repaired and checked by reading, not compiled, not run on hardware
//
#include <hip/hip_runtime.h>

typedef __attribute__((ext_vector_type(16))) __bf16 v16bf;
typedef __attribute__((ext_vector_type(8)))  __bf16 v8bf;
typedef __attribute__((ext_vector_type(8)))  float  v8f;
typedef __attribute__((ext_vector_type(4)))  int    v4i;

__device__ __forceinline__ v8f wmma_bf16(v16bf a, v16bf b, v8f c) {
  return __builtin_amdgcn_wmma_f32_16x16x32_bf16(false, a, false, b, (short)0, c,
                                                 false, false);
}

// A-matrix fragment (16x32 bf16): lane row = lane%16, group gh = lane/16.
// halves: K = 16*gh + [0..7] and K = 16*gh + 8 + [0..7]  (contiguous 16B chunks)
__device__ __forceinline__ v16bf load_a_bf16(const __bf16* rowk, int gh) {
  v8bf lo = *(const v8bf*)(rowk + 16 * gh);
  v8bf hi = *(const v8bf*)(rowk + 16 * gh + 8);
  v16bf r;
#pragma unroll
  for (int i = 0; i < 8; i++) { r[i] = lo[i]; r[8 + i] = hi[i]; }
  return r;
}

// ---- 16B global->LDS copy: async (ASYNCcnt) if available, else via VGPRs ----
#define HAS_ASYNC_LDS __has_builtin(__builtin_amdgcn_global_load_async_to_lds_b128)

__device__ __forceinline__ void cp16_g2l(const __bf16* g, __bf16* l) {
#if HAS_ASYNC_LDS
  typedef __attribute__((address_space(1))) v4i* gp_t;   // global int4*
  typedef __attribute__((address_space(3))) v4i* lp_t;   // LDS int4*
  __builtin_amdgcn_global_load_async_to_lds_b128((gp_t)g, (lp_t)l, 0, 0);
#else
  *(uint4*)l = *(const uint4*)g;
#endif
}

__device__ __forceinline__ void wait_async_done() {
#if HAS_ASYNC_LDS
#if __has_builtin(__builtin_amdgcn_s_wait_asynccnt)
  __builtin_amdgcn_s_wait_asynccnt(0);
#else
  asm volatile("s_wait_asynccnt 0" ::: "memory");
#endif
#endif
}

// ---------------- one-shot conversion / transpose kernels --------------------
__global__ __launch_bounds__(256) void cvt_x_kernel(const float* __restrict__ X,
                                                    __bf16* __restrict__ Y) {
  int i = (blockIdx.x * 256 + threadIdx.x) * 4;
  float4 v = *(const float4*)(X + i);
  Y[i + 0] = (__bf16)v.x; Y[i + 1] = (__bf16)v.y;
  Y[i + 2] = (__bf16)v.z; Y[i + 3] = (__bf16)v.w;
}

// Wh[h][64][128] fp32 -> WhT[h][128][64] bf16
__global__ __launch_bounds__(256) void cvt_whT_kernel(const float* __restrict__ W,
                                                      __bf16* __restrict__ WT) {
  int i = blockIdx.x * 256 + threadIdx.x;          // 0..65535
  int k = i & 63, d = (i >> 6) & 127, h = i >> 13;
  WT[((size_t)h * 128 + d) * 64 + k] = (__bf16)W[((size_t)h * 64 + k) * 128 + d];
}

// W_out[1024][64] fp32 -> WoutT[64][1024] bf16
__global__ __launch_bounds__(256) void cvt_woutT_kernel(const float* __restrict__ W,
                                                        __bf16* __restrict__ WT) {
  int i = blockIdx.x * 256 + threadIdx.x;          // 0..65535
  int k = i & 1023, n = i >> 10;
  WT[(size_t)n * 1024 + k] = (__bf16)W[(size_t)k * 64 + n];
}

// ---------------- layer-1 projection: xbf @ WhT -> hfeat + hfeatT ------------
__global__ __launch_bounds__(256) void proj1_kernel(
    const __bf16* __restrict__ X /*[B][N][64]*/,
    const __bf16* __restrict__ WhT /*[H][128][64]*/,
    __bf16* __restrict__ Y /*[B*H][N][128]*/,
    __bf16* __restrict__ YT /*[B*H][128][N]*/) {
  constexpr int N = 1024, CIN = 64, DD = 128, H = 8;
  int wave = threadIdx.x >> 5, lane = threadIdx.x & 31;
  int gw = blockIdx.x * (blockDim.x >> 5) + wave;
  int nt = gw & 7;  gw >>= 3;
  int mg = gw & 15; gw >>= 4;
  int b = gw >> 3, h = gw & 7;
  int gl = lane & 15, gh = lane >> 4;
  const __bf16* Xb = X + (size_t)b * N * CIN;
  const __bf16* Wc = WhT + ((size_t)h * DD + nt * 16 + gl) * CIN;
  v8f acc[4] = {};
#pragma unroll
  for (int ks = 0; ks < CIN; ks += 32) {
    v16bf bfr = *(const v16bf*)(Wc + ks + 16 * gh);
#pragma unroll
    for (int mt = 0; mt < 4; mt++) {
      int row = mg * 64 + mt * 16 + gl;
      v16bf af = load_a_bf16(Xb + (size_t)row * CIN + ks, gh);
      acc[mt] = wmma_bf16(af, bfr, acc[mt]);
    }
  }
  int g = b * H + h;
  __bf16* Yp  = Y + (size_t)g * N * DD;
  __bf16* YTp = YT + (size_t)g * DD * N;
#pragma unroll
  for (int mt = 0; mt < 4; mt++)
#pragma unroll
    for (int r = 0; r < 8; r++) {
      int m = mg * 64 + mt * 16 + r + 8 * gh;
      int col = nt * 16 + gl;
      __bf16 v = (__bf16)acc[mt][r];
      Yp[(size_t)m * DD + col] = v;
      YTp[(size_t)col * N + m] = v;
    }
}

// ---------------- layer-2 projection: cat @ WoutT -> h2 + h2T ----------------
__global__ __launch_bounds__(256) void proj2_kernel(
    const __bf16* __restrict__ X /*[B][N][1024]*/,
    const __bf16* __restrict__ WT /*[64][1024]*/,
    __bf16* __restrict__ Y /*[B][N][64]*/,
    __bf16* __restrict__ YT /*[B][64][N]*/) {
  constexpr int N = 1024, K = 1024, DO = 64;
  int wave = threadIdx.x >> 5, lane = threadIdx.x & 31;
  int gw = blockIdx.x * (blockDim.x >> 5) + wave;
  int nt = gw & 3;  gw >>= 2;
  int mg = gw & 15; gw >>= 4;
  int b = gw;
  int gl = lane & 15, gh = lane >> 4;
  const __bf16* Xb = X + (size_t)b * N * K;
  const __bf16* Wc = WT + (size_t)(nt * 16 + gl) * K;
  v8f acc[4] = {};
  for (int ks = 0; ks < K; ks += 32) {
    v16bf bfr = *(const v16bf*)(Wc + ks + 16 * gh);
#pragma unroll
    for (int mt = 0; mt < 4; mt++) {
      v16bf af = load_a_bf16(Xb + (size_t)(mg * 64 + mt * 16 + gl) * K + ks, gh);
      acc[mt] = wmma_bf16(af, bfr, acc[mt]);
    }
  }
  __bf16* Yp  = Y + (size_t)b * N * DO;
  __bf16* YTp = YT + (size_t)b * DO * N;
#pragma unroll
  for (int mt = 0; mt < 4; mt++)
#pragma unroll
    for (int r = 0; r < 8; r++) {
      int m = mg * 64 + mt * 16 + r + 8 * gh;
      int col = nt * 16 + gl;
      __bf16 v = (__bf16)acc[mt][r];
      Yp[(size_t)m * DO + col] = v;
      YTp[(size_t)col * N + m] = v;
    }
}

// ---- cooperative staging of one 32-key K/V tile into LDS (128 threads) -----
template <int D>
__device__ __forceinline__ void stage_tiles(__bf16* kdst, __bf16* vdst,
                                            const __bf16* fm, const __bf16* fmt,
                                            int kb, int tid) {
  constexpr int CH = 4 * D;  // number of 16B chunks per 32xD tile
  // K tile: rows kb..kb+31 of feat are one contiguous 64*D-byte block
#pragma unroll
  for (int ci = tid; ci < CH; ci += 128)
    cp16_g2l(fm + (size_t)kb * D + ci * 8, kdst + ci * 8);
  // V tile: D rows of featT, 32 keys (64B) each
#pragma unroll
  for (int ci = tid; ci < CH; ci += 128) {
    int d = ci >> 2, c = ci & 3;
    cp16_g2l(fmt + (size_t)d * 1024 + kb + c * 8, vdst + d * 32 + c * 8);
  }
}

// ---------------- fused masked-softmax attention (flash style) ----------------
// feat: [G][N][D] (Q,K), featT: [G][D][N] (V). 4 waves/block share one (b,h);
// K/V tiles double-buffered in LDS via async global->LDS copies.
template <int D, bool CAT_OUT>
__global__ __launch_bounds__(128) void attn_kernel(
    const __bf16* __restrict__ feat, const __bf16* __restrict__ featT,
    const float* __restrict__ graph, const float* __restrict__ bias,
    void* __restrict__ outp, int H) {
  constexpr int NN = 1024;
  constexpr int KC = D / 32;
  constexpr int NT = D / 16;
  __shared__ __bf16 kbuf[2][32 * D];
  __shared__ __bf16 vbuf[2][32 * D];
  __shared__ __bf16 plds[4][16 * 32];
  int tid = threadIdx.x;
  int wave = tid >> 5, lane = tid & 31;
  int gwave = blockIdx.x * 4 + wave;
  int qtile = gwave & 63, g = gwave >> 6;   // 4 waves/block: same g, adj qtiles
  int q0 = qtile * 16;
  int gl = lane & 15, gh = lane >> 4;
  const __bf16* fm  = feat  + (size_t)g * NN * D;
  const __bf16* fmt = featT + (size_t)g * D * NN;
  const float* bias_h = bias + (size_t)(g % H) * D;

  // Q fragments (rows q0..q0+15), A layout, straight from global
  v16bf qf[KC];
  const __bf16* qrow = fm + (size_t)(q0 + gl) * D;
#pragma unroll
  for (int c = 0; c < KC; c++) qf[c] = load_a_bf16(qrow + c * 32, gh);

  v8f O[NT] = {};
  float mrow[8], lsum[8];
#pragma unroll
  for (int r = 0; r < 8; r++) { mrow[r] = -3.0e38f; lsum[r] = 0.f; }

  __bf16* myp = plds[wave];

  stage_tiles<D>(kbuf[0], vbuf[0], fm, fmt, 0, tid);
  wait_async_done();
  __syncthreads();

  for (int kb = 0; kb < NN; kb += 32) {
    int cur = (kb >> 5) & 1;
    if (kb + 32 < NN)   // issue next tile's copies; overlap with compute below
      stage_tiles<D>(kbuf[cur ^ 1], vbuf[cur ^ 1], fm, fmt, kb + 32, tid);
    if (kb + 32 < NN) __builtin_prefetch(graph + (size_t)q0 * NN + kb + 32, 0, 1);
    const __bf16* kl = kbuf[cur];
    const __bf16* vl = vbuf[cur];

    // S = Q @ K^T for 32 keys (two 16x16 tiles), K fragments from LDS
    v8f s[2];
#pragma unroll
    for (int st = 0; st < 2; st++) {
      v8f a = {};
#pragma unroll
      for (int c = 0; c < KC; c++) {
        v16bf bf = *(const v16bf*)(kl + (size_t)(st * 16 + gl) * D + c * 32 + 16 * gh);
        a = wmma_bf16(qf[c], bf, a);
      }
      s[st] = a;
    }
    // mask exactly as reference: s *= graph; s == 0 -> -1e16
#pragma unroll
    for (int st = 0; st < 2; st++)
#pragma unroll
      for (int r = 0; r < 8; r++) {
        int m = q0 + r + 8 * gh;
        float gv = graph[(size_t)m * NN + kb + st * 16 + gl];
        float sv = s[st][r] * gv;
        s[st][r] = (sv == 0.f) ? -1.0e16f : sv;
      }
    // online softmax (row lives on 16 lanes; xor<16 stays in the half)
#pragma unroll
    for (int r = 0; r < 8; r++) {
      float v = fmaxf(s[0][r], s[1][r]);
#pragma unroll
      for (int off = 1; off < 16; off <<= 1) v = fmaxf(v, __shfl_xor(v, off, 32));
      float mn = fmaxf(mrow[r], v);
      float sc = __expf(mrow[r] - mn);
      mrow[r] = mn;
      float p0 = __expf(s[0][r] - mn);
      float p1 = __expf(s[1][r] - mn);
      s[0][r] = p0; s[1][r] = p1;
      float rs = p0 + p1;
#pragma unroll
      for (int off = 1; off < 16; off <<= 1) rs += __shfl_xor(rs, off, 32);
      lsum[r] = lsum[r] * sc + rs;
#pragma unroll
      for (int t = 0; t < NT; t++) O[t][r] *= sc;
    }
    // re-layout P (D layout -> A layout) through per-wave LDS tile
#pragma unroll
    for (int st = 0; st < 2; st++)
#pragma unroll
      for (int r = 0; r < 8; r++)
        myp[(r + 8 * gh) * 32 + st * 16 + gl] = (__bf16)s[st][r];
    v16bf pa = load_a_bf16(myp + gl * 32, gh);
    // O += P @ V ; V B-fragment: lane col d, 16 consecutive keys from LDS
#pragma unroll
    for (int t = 0; t < NT; t++) {
      v16bf vb = *(const v16bf*)(vl + (size_t)(t * 16 + gl) * 32 + 16 * gh);
      O[t] = wmma_bf16(pa, vb, O[t]);
    }
    wait_async_done();   // own async copies for next tile have landed
    __syncthreads();     // all waves' copies visible; reads of cur retired
  }
  // epilogue: normalize, + bias, leaky ReLU, store
#pragma unroll
  for (int t = 0; t < NT; t++) {
    float bv = bias_h[t * 16 + gl];
#pragma unroll
    for (int r = 0; r < 8; r++) {
      int m = q0 + r + 8 * gh;
      float v = O[t][r] / lsum[r] + bv;
      v = (v > 0.f) ? v : 0.01f * v;
      if constexpr (CAT_OUT) {
        int b = g / H, h = g % H;
        __bf16* op = (__bf16*)outp;
        op[((size_t)b * NN + m) * (size_t)(H * D) + (size_t)h * D + t * 16 + gl] =
            (__bf16)v;
      } else {
        float* op = (float*)outp;
        op[((size_t)g * NN + m) * D + t * 16 + gl] = v;
      }
    }
  }
}

extern "C" void kernel_launch(void* const* d_in, const int* in_sizes, int n_in,
                              void* d_out, int out_size, void* d_ws, size_t ws_size,
                              hipStream_t stream) {
  const float* flow_x = (const float*)d_in[0];  // [16,1024,64]
  const float* graph  = (const float*)d_in[1];  // [1024,1024]
  const float* Wh     = (const float*)d_in[2];  // [8,64,128]
  const float* bh     = (const float*)d_in[3];  // [8,128]
  const float* W_out  = (const float*)d_in[4];  // [1024,64]
  const float* b_out  = (const float*)d_in[5];  // [64]
  constexpr int B = 16, N = 1024, H = 8, D1 = 128, D2 = 64;

  __bf16* xbf    = (__bf16*)d_ws;                      // [B][N][64]        2 MB
  __bf16* whT    = xbf    + (size_t)B * N * 64;        // [H][128][64]    128 KB
  __bf16* woutT  = whT    + (size_t)H * 128 * 64;      // [64][1024]      128 KB
  __bf16* hfeat  = woutT  + (size_t)64 * 1024;         // [B*H][N][128]    32 MB
  __bf16* hfeatT = hfeat  + (size_t)B * H * N * D1;    // [B*H][128][N]    32 MB
  __bf16* cat    = hfeatT + (size_t)B * H * D1 * N;    // [B][N][1024]     32 MB
  __bf16* h2     = cat    + (size_t)B * N * (H * D1);  // [B][N][64]        2 MB
  __bf16* h2T    = h2     + (size_t)B * N * D2;        // [B][64][N]        2 MB

  cvt_x_kernel<<<(B * N * 64 / 4) / 256, 256, 0, stream>>>(flow_x, xbf);
  cvt_whT_kernel<<<(H * 128 * 64) / 256, 256, 0, stream>>>(Wh, whT);
  cvt_woutT_kernel<<<(64 * 1024) / 256, 256, 0, stream>>>(W_out, woutT);

  proj1_kernel<<<2048, 256, 0, stream>>>(xbf, whT, hfeat, hfeatT);
  attn_kernel<128, true><<<2048, 128, 0, stream>>>(hfeat, hfeatT, graph, bh, cat, H);
  proj2_kernel<<<128, 256, 0, stream>>>(cat, woutT, h2, h2T);
  attn_kernel<64, false><<<256, 128, 0, stream>>>(h2, h2T, graph, b_out, d_out, 1);
}
